// CrossModalAttention_32298154065925
// MI455X (gfx1250) — compile-verified, matrix operand order checked by reading, not run on hardware
//
#include <hip/hip_runtime.h>
#include <hip/hip_bf16.h>

#define E_DIM 1024
#define B_DIM 16
#define SV_DIM 576
#define SL_DIM 448
#define S_DIM 1024  // SV + SL

typedef __attribute__((ext_vector_type(16))) __bf16 v16bf;
typedef __attribute__((ext_vector_type(8)))  float  v8f;

union FragU { uint4 u[2]; v16bf v; };

__device__ __forceinline__ unsigned short f32_to_bf16_rne(float f) {
    union { float f; unsigned int u; } cv; cv.f = f;
    unsigned int u = cv.u;
    u += 0x7FFFu + ((u >> 16) & 1u);   // round to nearest even
    return (unsigned short)(u >> 16);
}

// ---------------------------------------------------------------------------
// Fragment fetch at a precomputed per-lane row pointer (already includes the
// lane-half K-offset). Layout: lanes 0-15 hold K=[0..7],[16..23], lanes 16-31
// hold K=[8..15],[24..31] -> two 16-byte loads at +0B and +32B.
// ---------------------------------------------------------------------------
__device__ __forceinline__ v16bf frag_at(const unsigned short* __restrict__ p) {
    FragU f;
    f.u[0] = ((const uint4*)p)[0];   // K = kbase+0 .. kbase+7
    f.u[1] = ((const uint4*)p)[2];   // K = kbase+16 .. kbase+23
    return f.v;
}

// ---------------------------------------------------------------------------
// One wave computes a (16*MW) x (16*NW) output tile: D = A(M,K) * B(N,K)^T,
// bf16 in, f32 accumulate via v_wmma_f32_16x16x32_bf16.
// K is a compile-time constant: the k-loop fully unrolls so every global load
// uses an immediate offset from MW+NW loop-invariant row pointers, and the
// scheduler can pipeline loads across k-steps.
// ---------------------------------------------------------------------------
template <int MW, int NW, int K>
__device__ __forceinline__ void wmma_gemm_nt(
    const unsigned short* __restrict__ A, int lda,
    const unsigned short* __restrict__ B, int ldb,
    int M0, int N0, int lane,
    v8f (&acc)[MW][NW])
{
    const int rsel = lane & 15;
    const int half = (lane >> 4) & 1;
    const unsigned short* pa[MW];
    const unsigned short* pb[NW];
#pragma unroll
    for (int i = 0; i < MW; ++i)
        pa[i] = A + (size_t)(M0 + 16 * i + rsel) * lda + half * 8;
#pragma unroll
    for (int j = 0; j < NW; ++j)
        pb[j] = B + (size_t)(N0 + 16 * j + rsel) * ldb + half * 8;

#pragma unroll
    for (int k = 0; k < K; k += 32) {
        v16bf afr[MW];
        v16bf bfr[NW];
#pragma unroll
        for (int i = 0; i < MW; ++i)
            afr[i] = frag_at(pa[i] + k);
#pragma unroll
        for (int j = 0; j < NW; ++j)
            bfr[j] = frag_at(pb[j] + k);
#pragma unroll
        for (int i = 0; i < MW; ++i)
#pragma unroll
            for (int j = 0; j < NW; ++j)
                acc[i][j] = __builtin_amdgcn_wmma_f32_16x16x32_bf16(
                    false, afr[i], false, bfr[j], (short)0, acc[i][j], false, false);
    }
}

// C/D 16x16 f32 layout: VGPR r -> M = r + 8*(lane>>4), N = lane&15
__device__ __forceinline__ void store_tile_bf16(unsigned short* __restrict__ D, int ldd,
                                                int M0, int N0, int lane, v8f c) {
    const int m = M0 + ((lane >> 4) & 1) * 8;
    const int n = N0 + (lane & 15);
#pragma unroll
    for (int r = 0; r < 8; ++r)
        D[(size_t)(m + r) * ldd + n] = f32_to_bf16_rne(c[r]);
}

__device__ __forceinline__ void store_tile_bf16_T(unsigned short* __restrict__ DT, int ldt,
                                                  int M0, int N0, int lane, v8f c) {
    const int m = M0 + ((lane >> 4) & 1) * 8;
    const int n = N0 + (lane & 15);
#pragma unroll
    for (int r = 0; r < 8; ++r)
        DT[(size_t)n * ldt + (m + r)] = f32_to_bf16_rne(c[r]);
}

__device__ __forceinline__ void store_tile_f32_scaled(float* __restrict__ D, int ldd,
                                                      int M0, int N0, int lane, v8f c, float s) {
    const int m = M0 + ((lane >> 4) & 1) * 8;
    const int n = N0 + (lane & 15);
#pragma unroll
    for (int r = 0; r < 8; ++r)
        D[(size_t)(m + r) * ldd + n] = c[r] * s;
}

__device__ __forceinline__ void store_tile_f32_bias(float* __restrict__ D, int ldd,
                                                    int M0, int N0, int lane, v8f c,
                                                    const float* __restrict__ bias) {
    const int m = M0 + ((lane >> 4) & 1) * 8;
    const int n = N0 + (lane & 15);
    const float bv = bias[n];
#pragma unroll
    for (int r = 0; r < 8; ++r)
        D[(size_t)(m + r) * ldd + n] = c[r] + bv;
}

// Wave tile geometry: 8 waves per block, 2 (M) x 4 (N); wave = 64x32 output.
#define MW_T 4
#define NW_T 2
#define BLK_M 128
#define BLK_N 128

__device__ __forceinline__ void wave_tile_origin(int& M0, int& N0, int& lane) {
    lane = threadIdx.x & 31;
    const int wid = threadIdx.x >> 5;
    M0 = blockIdx.y * BLK_M + (wid >> 2) * 64;   // 2 wave rows
    N0 = blockIdx.x * BLK_N + (wid & 3) * 32;    // 4 wave cols
}

// ---------------------------------------------------------------------------
// Phase 0: f32 -> bf16 packing
// ---------------------------------------------------------------------------
__global__ void cma_pack_tokens(const float* __restrict__ vis,
                                const float* __restrict__ lang,
                                unsigned short* __restrict__ tok) {
    size_t idx = (size_t)blockIdx.x * blockDim.x + threadIdx.x;
    const size_t total = (size_t)B_DIM * S_DIM * E_DIM;
    if (idx >= total) return;
    const int e = (int)(idx % E_DIM);
    const int s = (int)((idx / E_DIM) % S_DIM);
    const int b = (int)(idx / ((size_t)E_DIM * S_DIM));
    const float v = (s < SV_DIM)
        ? vis[((size_t)b * SV_DIM + s) * E_DIM + e]
        : lang[((size_t)b * SL_DIM + (s - SV_DIM)) * E_DIM + e];
    tok[idx] = f32_to_bf16_rne(v);
}

__global__ void cma_pack_weights(const float* __restrict__ Wv,
                                 const float* __restrict__ Wl,
                                 const float* __restrict__ Wo,
                                 unsigned short* __restrict__ Wbf) {
    size_t idx = (size_t)blockIdx.x * blockDim.x + threadIdx.x;
    const size_t EE = (size_t)E_DIM * E_DIM;
    if (idx >= 3 * EE) return;
    float v;
    if (idx < EE)           v = Wv[idx];
    else if (idx < 2 * EE)  v = Wl[idx - EE];
    else                    v = Wo[idx - 2 * EE];
    Wbf[idx] = f32_to_bf16_rne(v);
}

// ---------------------------------------------------------------------------
// Phase 1: projection  x[b,s,:] = tok[b,s,:] @ W^T   (W = Wv for s<SV else Wl)
// writes x (row-major bf16) and xT (transposed bf16)
// ---------------------------------------------------------------------------
__global__ void cma_proj_kernel(const unsigned short* __restrict__ tok,
                                const unsigned short* __restrict__ Wv,
                                const unsigned short* __restrict__ Wl,
                                unsigned short* __restrict__ x,
                                unsigned short* __restrict__ xT) {
    const int b = blockIdx.z;
    int M0, N0, lane;
    wave_tile_origin(M0, N0, lane);
    const unsigned short* A = tok + (size_t)b * S_DIM * E_DIM;
    const unsigned short* W = (M0 < SV_DIM) ? Wv : Wl;   // SV % 64 == 0: no straddle
    v8f acc[MW_T][NW_T] = {};
    wmma_gemm_nt<MW_T, NW_T, E_DIM>(A, E_DIM, W, E_DIM, M0, N0, lane, acc);
    unsigned short* xb  = x  + (size_t)b * S_DIM * E_DIM;
    unsigned short* xTb = xT + (size_t)b * E_DIM * S_DIM;
#pragma unroll
    for (int i = 0; i < MW_T; ++i)
#pragma unroll
        for (int j = 0; j < NW_T; ++j) {
            store_tile_bf16(xb, E_DIM, M0 + 16 * i, N0 + 16 * j, lane, acc[i][j]);
            store_tile_bf16_T(xTb, S_DIM, M0 + 16 * i, N0 + 16 * j, lane, acc[i][j]);
        }
}

// ---------------------------------------------------------------------------
// Phase 2: scores[b,q,k] = (x[b,q,:] . x[b,k,:]) * E^-0.5     (f32 out)
// ---------------------------------------------------------------------------
__global__ void cma_scores_kernel(const unsigned short* __restrict__ x,
                                  float* __restrict__ scores) {
    const int b = blockIdx.z;
    int M0, N0, lane;
    wave_tile_origin(M0, N0, lane);
    const unsigned short* xb = x + (size_t)b * S_DIM * E_DIM;
    v8f acc[MW_T][NW_T] = {};
    wmma_gemm_nt<MW_T, NW_T, E_DIM>(xb, E_DIM, xb, E_DIM, M0, N0, lane, acc);
    float* sb = scores + (size_t)b * S_DIM * S_DIM;
    const float scale = 0.03125f;   // 1024^-0.5
#pragma unroll
    for (int i = 0; i < MW_T; ++i)
#pragma unroll
        for (int j = 0; j < NW_T; ++j)
            store_tile_f32_scaled(sb, S_DIM, M0 + 16 * i, N0 + 16 * j, lane, acc[i][j], scale);
}

// ---------------------------------------------------------------------------
// Phase 3: row softmax (f32 in, bf16 probabilities out). One block per row.
// ---------------------------------------------------------------------------
__global__ void cma_softmax_kernel(const float* __restrict__ scores,
                                   unsigned short* __restrict__ attn) {
    __shared__ float red[256];
    const int q = blockIdx.x;
    const int b = blockIdx.y;
    const float* row = scores + ((size_t)b * S_DIM + q) * S_DIM;
    unsigned short* arow = attn + ((size_t)b * S_DIM + q) * S_DIM;
    const int t = threadIdx.x;

    float v[4];
    float m = -INFINITY;
#pragma unroll
    for (int i = 0; i < 4; ++i) {
        v[i] = row[t + i * 256];
        m = fmaxf(m, v[i]);
    }
    red[t] = m;
    __syncthreads();
    for (int s = 128; s > 0; s >>= 1) {
        if (t < s) red[t] = fmaxf(red[t], red[t + s]);
        __syncthreads();
    }
    const float rmax = red[0];
    __syncthreads();

    float sum = 0.0f;
#pragma unroll
    for (int i = 0; i < 4; ++i) {
        v[i] = __expf(v[i] - rmax);
        sum += v[i];
    }
    red[t] = sum;
    __syncthreads();
    for (int s = 128; s > 0; s >>= 1) {
        if (t < s) red[t] += red[t + s];
        __syncthreads();
    }
    const float inv = 1.0f / red[0];
#pragma unroll
    for (int i = 0; i < 4; ++i)
        arow[t + i * 256] = f32_to_bf16_rne(v[i] * inv);
}

// ---------------------------------------------------------------------------
// Phase 4: attended[b,q,e] = sum_k attn[b,q,k] * x[b,k,e]   (B operand = xT rows)
// ---------------------------------------------------------------------------
__global__ void cma_attended_kernel(const unsigned short* __restrict__ attn,
                                    const unsigned short* __restrict__ xT,
                                    unsigned short* __restrict__ att) {
    const int b = blockIdx.z;
    int M0, N0, lane;
    wave_tile_origin(M0, N0, lane);
    const unsigned short* A  = attn + (size_t)b * S_DIM * S_DIM;
    const unsigned short* Bm = xT   + (size_t)b * E_DIM * S_DIM;
    v8f acc[MW_T][NW_T] = {};
    wmma_gemm_nt<MW_T, NW_T, S_DIM>(A, S_DIM, Bm, S_DIM, M0, N0, lane, acc);
    unsigned short* ab = att + (size_t)b * S_DIM * E_DIM;
#pragma unroll
    for (int i = 0; i < MW_T; ++i)
#pragma unroll
        for (int j = 0; j < NW_T; ++j)
            store_tile_bf16(ab, E_DIM, M0 + 16 * i, N0 + 16 * j, lane, acc[i][j]);
}

// ---------------------------------------------------------------------------
// Phase 5: out[b,s,f] = attended[b,s,:] . Wo[f,:] + bo[f]   (f32 out)
// ---------------------------------------------------------------------------
__global__ void cma_out_kernel(const unsigned short* __restrict__ att,
                               const unsigned short* __restrict__ Wo,
                               const float* __restrict__ bo,
                               float* __restrict__ out) {
    const int b = blockIdx.z;
    int M0, N0, lane;
    wave_tile_origin(M0, N0, lane);
    const unsigned short* A = att + (size_t)b * S_DIM * E_DIM;
    v8f acc[MW_T][NW_T] = {};
    wmma_gemm_nt<MW_T, NW_T, E_DIM>(A, E_DIM, Wo, E_DIM, M0, N0, lane, acc);
    float* ob = out + (size_t)b * S_DIM * E_DIM;
#pragma unroll
    for (int i = 0; i < MW_T; ++i)
#pragma unroll
        for (int j = 0; j < NW_T; ++j)
            store_tile_f32_bias(ob, E_DIM, M0 + 16 * i, N0 + 16 * j, lane, acc[i][j], bo);
}

// ---------------------------------------------------------------------------
extern "C" void kernel_launch(void* const* d_in, const int* in_sizes, int n_in,
                              void* d_out, int out_size, void* d_ws, size_t ws_size,
                              hipStream_t stream) {
    (void)in_sizes; (void)n_in; (void)out_size; (void)ws_size;
    const float* vis  = (const float*)d_in[0];
    const float* lang = (const float*)d_in[1];
    const float* Wv   = (const float*)d_in[2];
    const float* Wl   = (const float*)d_in[3];
    const float* Wo   = (const float*)d_in[4];
    const float* bo   = (const float*)d_in[5];
    float* out = (float*)d_out;

    const size_t EE2  = (size_t)E_DIM * E_DIM * sizeof(unsigned short);          // 2 MB
    const size_t BSE2 = (size_t)B_DIM * S_DIM * E_DIM * sizeof(unsigned short);  // 32 MB
    const size_t BSS4 = (size_t)B_DIM * S_DIM * S_DIM * sizeof(float);           // 64 MB

    unsigned char* p = (unsigned char*)d_ws;
    unsigned short* Wbf    = (unsigned short*)p; p += 3 * EE2;
    unsigned short* tok    = (unsigned short*)p; p += BSE2;
    unsigned short* xbuf   = (unsigned short*)p; p += BSE2;
    unsigned short* xTbuf  = (unsigned short*)p; p += BSE2;
    float*          scores = (float*)p;          p += BSS4;
    unsigned short* attn   = (unsigned short*)p; p += BSE2;  // B*S*S*2 (S==E)
    unsigned short* att    = (unsigned short*)p; p += BSE2;

    const unsigned short* Wv_bf = Wbf;
    const unsigned short* Wl_bf = Wbf + (size_t)E_DIM * E_DIM;
    const unsigned short* Wo_bf = Wbf + 2 * (size_t)E_DIM * E_DIM;

    // Phase 0: pack to bf16
    {
        const size_t totTok = (size_t)B_DIM * S_DIM * E_DIM;
        cma_pack_tokens<<<dim3((unsigned)((totTok + 255) / 256)), dim3(256), 0, stream>>>(vis, lang, tok);
        const size_t totW = 3 * (size_t)E_DIM * E_DIM;
        cma_pack_weights<<<dim3((unsigned)((totW + 255) / 256)), dim3(256), 0, stream>>>(Wv, Wl, Wo, Wbf);
    }

    const dim3 gemmGrid(S_DIM / BLK_N, S_DIM / BLK_M, B_DIM);  // 8 x 8 x 16
    const dim3 gemmBlock(256);                                 // 8 waves, 128x128 block tile

    // Phase 1: projections (+ transposed copy of x)
    cma_proj_kernel<<<gemmGrid, gemmBlock, 0, stream>>>(tok, Wv_bf, Wl_bf, xbuf, xTbuf);

    // Phase 2: scaled scores
    cma_scores_kernel<<<gemmGrid, gemmBlock, 0, stream>>>(xbuf, scores);

    // Phase 3: softmax
    cma_softmax_kernel<<<dim3(S_DIM, B_DIM), dim3(256), 0, stream>>>(scores, attn);

    // Phase 4: attended = attn @ x
    cma_attended_kernel<<<gemmGrid, gemmBlock, 0, stream>>>(attn, xTbuf, att);

    // Phase 5: out projection + bias
    cma_out_kernel<<<gemmGrid, gemmBlock, 0, stream>>>(att, Wo_bf, bo, out);
}
